// NonLocalBlock_68212670595430
// MI455X (gfx1250) — matrix operand order repaired and checked by reading, not compiled, not run on hardware
//
#include <hip/hip_runtime.h>

typedef __attribute__((ext_vector_type(16))) _Float16 v16h;
typedef __attribute__((ext_vector_type(8)))  _Float16 v8h;
typedef __attribute__((ext_vector_type(2)))  _Float16 v2h;
typedef __attribute__((ext_vector_type(8)))  float    v8f;

#define C_IN 64
#define CH   32
#define DD   8
#define HH   48
#define WW   48
#define LTOT (DD * HH * WW)   // 18432
#define MTOT (LTOT / 4)       // 4608

// ---------------------------------------------------------------------------
// Kernel 1: theta projection, output thetaT [L][32] f16 (row-major per l).
// ---------------------------------------------------------------------------
__global__ __launch_bounds__(256) void theta_proj_kernel(
    const float* __restrict__ x, const float* __restrict__ w,
    const float* __restrict__ b, _Float16* __restrict__ thetaT) {
  __shared__ float ws[CH * C_IN];
  __shared__ float bs[CH];
  for (int i = threadIdx.x; i < CH * C_IN; i += blockDim.x) ws[i] = w[i];
  if (threadIdx.x < CH) bs[threadIdx.x] = b[threadIdx.x];
  __syncthreads();

  const int l = blockIdx.x * blockDim.x + threadIdx.x;  // L divisible by 256
  float acc[CH];
#pragma unroll
  for (int o = 0; o < CH; ++o) acc[o] = bs[o];
  for (int c = 0; c < C_IN; ++c) {
    const float xv = x[(size_t)c * LTOT + l];
#pragma unroll
    for (int o = 0; o < CH; ++o) acc[o] = fmaf(ws[o * C_IN + c], xv, acc[o]);
  }
  _Float16* dst = thetaT + (size_t)l * CH;
#pragma unroll
  for (int o = 0; o < CH; ++o) dst[o] = (_Float16)acc[o];
}

// ---------------------------------------------------------------------------
// Kernel 2: fused (1,2,2) max-pool + phi & g projections.
//   phiT [M][32] f16, gC [32][M] f16
// ---------------------------------------------------------------------------
__global__ __launch_bounds__(256) void pool_proj_kernel(
    const float* __restrict__ x,
    const float* __restrict__ w_phi, const float* __restrict__ b_phi,
    const float* __restrict__ w_g,   const float* __restrict__ b_g,
    _Float16* __restrict__ phiT, _Float16* __restrict__ gC) {
  __shared__ float wp[CH * C_IN];
  __shared__ float wg[CH * C_IN];
  __shared__ float bp[CH];
  __shared__ float bg[CH];
  for (int i = threadIdx.x; i < CH * C_IN; i += blockDim.x) {
    wp[i] = w_phi[i];
    wg[i] = w_g[i];
  }
  if (threadIdx.x < CH) {
    bp[threadIdx.x] = b_phi[threadIdx.x];
    bg[threadIdx.x] = b_g[threadIdx.x];
  }
  __syncthreads();

  const int m = blockIdx.x * blockDim.x + threadIdx.x;  // M divisible by 256
  const int d  = m / 576;
  const int rm = m % 576;
  const int hp = rm / 24;
  const int wq = rm % 24;
  const float* xb = x + (size_t)d * (HH * WW) + hp * (2 * WW) + wq * 2;

  float ap[CH], ag[CH];
#pragma unroll
  for (int o = 0; o < CH; ++o) { ap[o] = bp[o]; ag[o] = bg[o]; }

  for (int c = 0; c < C_IN; ++c) {
    const float* p = xb + (size_t)c * LTOT;
    float xv = fmaxf(fmaxf(p[0], p[1]), fmaxf(p[WW], p[WW + 1]));
#pragma unroll
    for (int o = 0; o < CH; ++o) {
      ap[o] = fmaf(wp[o * C_IN + c], xv, ap[o]);
      ag[o] = fmaf(wg[o * C_IN + c], xv, ag[o]);
    }
  }
  _Float16* pd = phiT + (size_t)m * CH;
#pragma unroll
  for (int o = 0; o < CH; ++o) {
    pd[o] = (_Float16)ap[o];
    gC[(size_t)o * MTOT + m] = (_Float16)ag[o];
  }
}

// ---------------------------------------------------------------------------
// Cross-half (lane ^ 16) exchange. gfx1250 is a gfx12-family target, so use
// v_permlanex16_b32 (pure VALU, co-executes with the WMMA pipe) with identity
// lane selects; keeps the DS pipe idle in the attention kernel.
// ---------------------------------------------------------------------------
__device__ __forceinline__ unsigned xhalf_u(unsigned v) {
#if __has_builtin(__builtin_amdgcn_permlanex16)
  int i = (int)v;
  i = __builtin_amdgcn_permlanex16(i, i, 0x76543210, 0xfedcba98, false, false);
  return (unsigned)i;
#else
  return (unsigned)__shfl_xor((int)v, 16, 32);
#endif
}
__device__ __forceinline__ float xhalf_f(float v) {
  return __builtin_bit_cast(float, xhalf_u(__builtin_bit_cast(unsigned, v)));
}

__device__ __forceinline__ v16h ldA(const _Float16* base, int half) {
  const v8h lo = *(const v8h*)(base + half * 16);
  const v8h hi = *(const v8h*)(base + half * 16 + 8);
  return __builtin_shufflevector(lo, hi, 0, 1, 2, 3, 4, 5, 6, 7,
                                 8, 9, 10, 11, 12, 13, 14, 15);
}

__device__ __forceinline__ v8f wmma_f16(v16h a, v16h b, v8f c) {
  return __builtin_amdgcn_wmma_f32_16x16x32_f16(false, a, false, b,
                                                (short)0, c, false, false);
}

__device__ __forceinline__ unsigned pack2(float lo, float hi) {
  v2h t = {(_Float16)lo, (_Float16)hi};
  return __builtin_bit_cast(unsigned, t);
}

// ---------------------------------------------------------------------------
// Kernel 3: fused flash-attention (online softmax over the M axis).
// Each wave owns 16 query columns and streams all M=4608 keys in steps of 32.
// Per step: 2 score WMMAs + softmax bookkeeping + 2 accumulate WMMAs.
// ---------------------------------------------------------------------------
__global__ __launch_bounds__(128) void attn_kernel(
    const _Float16* __restrict__ thetaT, const _Float16* __restrict__ phiT,
    const _Float16* __restrict__ gC, float* __restrict__ y_ws) {
  const int wave = threadIdx.x >> 5;
  const int lane = threadIdx.x & 31;
  const int half = lane >> 4;
  const int r    = lane & 15;
  const int l0   = blockIdx.x * 64 + wave * 16;

  // B fragment of theta: column l = l0+r, elements j = K = half*16 + j.
  const v16h bth = *(const v16h*)(thetaT + (size_t)(l0 + r) * CH + half * 16);

  v8f yacc0 = {};
  v8f yacc1 = {};
  float run_max = -__builtin_inff();
  float run_sum = 0.0f;

  for (int m0 = 0; m0 < MTOT; m0 += 32) {
    // ---- scores for 32 keys: two WMMAs (rows m0..+15, m0+16..+31) ----
    const v16h aphi0 = ldA(phiT + (size_t)(m0 + r) * CH, half);
    const v16h aphi1 = ldA(phiT + (size_t)(m0 + 16 + r) * CH, half);
    v8f zero = {};
    v8f s0 = wmma_f16(aphi0, bth, zero);
    v8f s1 = wmma_f16(aphi1, bth, zero);

    if (m0 + 32 < MTOT) {  // prefetch next key/value chunk
      __builtin_prefetch(phiT + (size_t)(m0 + 32 + r) * CH, 0, 1);
      __builtin_prefetch(gC + (size_t)r * MTOT + m0 + 32, 0, 1);
    }

    // ---- per-column max over 32 rows (16 local + 16 in partner lane) ----
    float cm = s0[0];
#pragma unroll
    for (int j = 1; j < 8; ++j) cm = fmaxf(cm, s0[j]);
#pragma unroll
    for (int j = 0; j < 8; ++j) cm = fmaxf(cm, s1[j]);
    cm = fmaxf(cm, xhalf_f(cm));

    const float nm    = fmaxf(run_max, cm);
    const float scale = __expf(run_max - nm);
    run_max = nm;

    float e0[8], e1[8];
    float lsum = 0.0f;
#pragma unroll
    for (int j = 0; j < 8; ++j) {
      e0[j] = __expf(s0[j] - nm);
      e1[j] = __expf(s1[j] - nm);
      lsum += e0[j] + e1[j];
    }
    lsum += xhalf_f(lsum);
    run_sum = run_sum * scale + lsum;

#pragma unroll
    for (int j = 0; j < 8; ++j) { yacc0[j] *= scale; yacc1[j] *= scale; }

    // ---- build B fragment of P = exp(scores - nm), 32x16 f16 ----
    // Pack to f16 pairs first, then exchange packed dwords (8 xhalf, not 16).
    unsigned pk0[4], pk1[4];
#pragma unroll
    for (int k = 0; k < 4; ++k) {
      pk0[k] = pack2(e0[2 * k], e0[2 * k + 1]);
      pk1[k] = pack2(e1[2 * k], e1[2 * k + 1]);
    }
    unsigned b2u[8];
#pragma unroll
    for (int k = 0; k < 4; ++k) {
      const unsigned x0 = xhalf_u(pk0[k]);  // partner s0 rows 8..15 (or 0..7)
      const unsigned x1 = xhalf_u(pk1[k]);  // partner s1 rows
      // half=0 lane: K 0..15  = local e0 | partner e0
      // half=1 lane: K 16..31 = partner e1 | local e1
      b2u[k]     = half ? x1 : pk0[k];
      b2u[k + 4] = half ? pk1[k] : x0;
    }
    typedef __attribute__((ext_vector_type(8))) unsigned v8u;
    const v16h b2 =
        __builtin_bit_cast(v16h, (v8u){b2u[0], b2u[1], b2u[2], b2u[3],
                                       b2u[4], b2u[5], b2u[6], b2u[7]});

    // ---- y[c, l] += g[c, m0+K] * P[K, l] : two WMMAs (c blocks) ----
    const v16h ag0 = ldA(gC + (size_t)r * MTOT + m0, half);
    const v16h ag1 = ldA(gC + (size_t)(16 + r) * MTOT + m0, half);
    yacc0 = wmma_f16(ag0, b2, yacc0);
    yacc1 = wmma_f16(ag1, b2, yacc1);
  }

  const float inv = __builtin_amdgcn_rcpf(run_sum);
  const int lcol = l0 + r;
#pragma unroll
  for (int v = 0; v < 8; ++v) {
    y_ws[(size_t)(half * 8 + v) * LTOT + lcol]      = yacc0[v] * inv;
    y_ws[(size_t)(16 + half * 8 + v) * LTOT + lcol] = yacc1[v] * inv;
  }
}

// ---------------------------------------------------------------------------
// Kernel 4: output projection h (32 -> 64) + bias + residual.
// ---------------------------------------------------------------------------
__global__ __launch_bounds__(256) void out_proj_kernel(
    const float* __restrict__ x, const float* __restrict__ w_h,
    const float* __restrict__ b_h, const float* __restrict__ y_ws,
    float* __restrict__ out) {
  __shared__ float wh[C_IN * CH];
  __shared__ float bh[C_IN];
  for (int i = threadIdx.x; i < C_IN * CH; i += blockDim.x) wh[i] = w_h[i];
  if (threadIdx.x < C_IN) bh[threadIdx.x] = b_h[threadIdx.x];
  __syncthreads();

  const int l = blockIdx.x * blockDim.x + threadIdx.x;
  float acc[C_IN];
#pragma unroll
  for (int c = 0; c < C_IN; ++c) acc[c] = bh[c];
  for (int ch = 0; ch < CH; ++ch) {
    const float yv = y_ws[(size_t)ch * LTOT + l];
#pragma unroll
    for (int c = 0; c < C_IN; ++c) acc[c] = fmaf(wh[c * CH + ch], yv, acc[c]);
  }
#pragma unroll
  for (int c = 0; c < C_IN; ++c)
    out[(size_t)c * LTOT + l] = acc[c] + x[(size_t)c * LTOT + l];
}

// ---------------------------------------------------------------------------
extern "C" void kernel_launch(void* const* d_in, const int* in_sizes, int n_in,
                              void* d_out, int out_size, void* d_ws, size_t ws_size,
                              hipStream_t stream) {
  (void)in_sizes; (void)n_in; (void)out_size; (void)ws_size;
  const float* x       = (const float*)d_in[0];
  const float* w_theta = (const float*)d_in[1];
  const float* b_theta = (const float*)d_in[2];
  const float* w_phi   = (const float*)d_in[3];
  const float* b_phi   = (const float*)d_in[4];
  const float* w_g     = (const float*)d_in[5];
  const float* b_g     = (const float*)d_in[6];
  const float* w_h     = (const float*)d_in[7];
  const float* b_h     = (const float*)d_in[8];

  char* ws = (char*)d_ws;
  const size_t thetaT_bytes = (size_t)LTOT * CH * sizeof(_Float16);  // 1179648
  const size_t phiT_bytes   = (size_t)MTOT * CH * sizeof(_Float16);  // 294912
  const size_t gC_bytes     = (size_t)CH * MTOT * sizeof(_Float16);  // 294912
  _Float16* thetaT = (_Float16*)ws;
  _Float16* phiT   = (_Float16*)(ws + thetaT_bytes);
  _Float16* gC     = (_Float16*)(ws + thetaT_bytes + phiT_bytes);
  float*    y_ws   = (float*)(ws + thetaT_bytes + phiT_bytes + gC_bytes);

  theta_proj_kernel<<<LTOT / 256, 256, 0, stream>>>(x, w_theta, b_theta, thetaT);
  pool_proj_kernel<<<MTOT / 256, 256, 0, stream>>>(x, w_phi, b_phi, w_g, b_g,
                                                   phiT, gC);
  attn_kernel<<<LTOT / 64, 128, 0, stream>>>(thetaT, phiT, gC, y_ws);
  out_proj_kernel<<<LTOT / 256, 256, 0, stream>>>(x, w_h, b_h, y_ws,
                                                  (float*)d_out);
}
